// AttentionLePE_14276471292629
// MI455X (gfx1250) — matrix operand order repaired and checked by reading, not compiled, not run on hardware
//
#include <hip/hip_runtime.h>

// ---------------------------------------------------------------------------
// AttentionLePE for MI455X (gfx1250), wave32 + WMMA f32_16x16x32_f16
// + async global->LDS (ASYNCcnt) for A tiles
// + Tensor Data Mover (TENSORcnt) for weight tiles, with LDS padding
// B=8, H=W=32 (N=1024), C=512, HEADS=16, hd=32
// ---------------------------------------------------------------------------

typedef __attribute__((ext_vector_type(16))) _Float16 v16h;
typedef __attribute__((ext_vector_type(8)))  _Float16 v8h;
typedef __attribute__((ext_vector_type(8)))  float    v8f;
typedef __attribute__((ext_vector_type(4)))  unsigned int v4u;
typedef __attribute__((ext_vector_type(8)))  int      v8i;
typedef __attribute__((ext_vector_type(4)))  int      v4i;

#define BATCH   8
#define NPOS    1024          // 32*32
#define CDIM    512
#define NHEADS  16
#define HD      32
#define MROWS   (BATCH * NPOS)   // 8192
#define QKV_N   (3 * CDIM)       // 1536

#ifndef __has_builtin
#define __has_builtin(x) 0
#endif
#if __has_builtin(__builtin_amdgcn_tensor_load_to_lds)
#define HAVE_TDM 1
#else
#define HAVE_TDM 0
#endif

__device__ __forceinline__ v8f zero8() {
  v8f z;
#pragma unroll
  for (int i = 0; i < 8; ++i) z[i] = 0.0f;
  return z;
}

// 16 contiguous halves -> v16h (two 16B loads, works for global & LDS ptrs)
__device__ __forceinline__ v16h frag_contig(const _Float16* p) {
  v8h a = *(const v8h*)p;
  v8h b = *(const v8h*)(p + 8);
  v16h r;
#pragma unroll
  for (int j = 0; j < 8; ++j) { r[j] = a[j]; r[j + 8] = b[j]; }
  return r;
}

// A-layout split: elements 0..7 = p[0..7], 8..15 = p[16..23]
__device__ __forceinline__ v16h frag_split(const _Float16* p) {
  v8h a = *(const v8h*)p;
  v8h b = *(const v8h*)(p + 16);
  v16h r;
#pragma unroll
  for (int j = 0; j < 8; ++j) { r[j] = a[j]; r[j + 8] = b[j]; }
  return r;
}

__device__ __forceinline__ v8f wmma_f16(v16h a, v16h b, v8f c) {
  return __builtin_amdgcn_wmma_f32_16x16x32_f16(false, a, false, b,
                                                (short)0, c, false, false);
}

// 32-bit LDS byte address of a shared-memory object (generic -> AS3 -> int)
__device__ __forceinline__ unsigned lds_addr_of(const void* p) {
  return (unsigned)(unsigned long long)(
      const __attribute__((address_space(3))) char*)p;
}

// Async copy engine: 16 bytes global -> LDS per lane (ASYNCcnt tracked)
__device__ __forceinline__ void async_copy_b128(unsigned lds_byte_addr,
                                                const void* gaddr) {
  asm volatile("global_load_async_to_lds_b128 %0, %1, off"
               :
               : "v"(lds_byte_addr), "v"(gaddr)
               : "memory");
}

__device__ __forceinline__ void wait_async0() {
  asm volatile("s_wait_asynccnt 0" ::: "memory");
}

__device__ __forceinline__ void wait_tensor0() {
#if __has_builtin(__builtin_amdgcn_s_wait_tensorcnt)
  __builtin_amdgcn_s_wait_tensorcnt(0);
#else
  asm volatile("s_wait_tensorcnt 0" ::: "memory");
#endif
}

#if HAVE_TDM
// TDM: load a 2D tile (tile_rows x 32 f16) global->LDS, one issue per wave.
// LDS dest gets 8 halves of padding per 32-half row (row stride 40 halves),
// via D#.pad_interval=3 (16 DWORDs) / pad_amount=3 (4 DWORDs).
__device__ __forceinline__ void tdm_load_tile_32xR(unsigned lds_byte,
                                                   const void* gsrc,
                                                   unsigned tile_rows,
                                                   unsigned row_stride_elems) {
  unsigned long long ga = (unsigned long long)gsrc;
  v4u g0;
  g0[0] = 1u;                                  // count=1, user descriptor
  g0[1] = lds_byte;                            // LDS byte address
  g0[2] = (unsigned)ga;                        // global_addr[31:0]
  g0[3] = (unsigned)((ga >> 32) & 0x01FFFFFFu) // global_addr[56:32]
          | (2u << 30);                        // type=2 ("image")
  v8i g1;
  g1[0] = (int)((1u << 16)      // data_size = 2 bytes
                | (1u << 20)    // pad_enable
                | (3u << 22)    // pad_interval: 16 DWORDs (one 32-half row)
                | (3u << 25));  // pad_amount: 4 DWORDs (8 halves)
  g1[1] = (int)(row_stride_elems << 16);   // tensor_dim0[15:0]
  g1[2] = (int)(tile_rows << 16);          // tensor_dim1[15:0]
  g1[3] = (int)(32u << 16);                // tile_dim0 = 32 elems
  g1[4] = (int)tile_rows;                  // tile_dim1 (tile_dim2 = 0)
  g1[5] = (int)row_stride_elems;           // tensor_dim0_stride[31:0]
  g1[6] = 0;
  g1[7] = 0;
  v4i gz = {0, 0, 0, 0};
#if __clang_major__ >= 23
  v8i gz8 = {0, 0, 0, 0, 0, 0, 0, 0};
  __builtin_amdgcn_tensor_load_to_lds(g0, g1, gz, gz, gz8, 0);
#else
  __builtin_amdgcn_tensor_load_to_lds(g0, g1, gz, gz, 0);
#endif
}
#endif

// ---------------------------------------------------------------------------
// fp32 -> f16 conversion
// ---------------------------------------------------------------------------
__global__ void cvt_f32_to_f16(const float* __restrict__ in,
                               _Float16* __restrict__ out, int n) {
  int i = blockIdx.x * 256 + threadIdx.x;
  if (i < n) out[i] = (_Float16)in[i];
}

// ---------------------------------------------------------------------------
// QKV GEMM: [8192,512] x [1536,512]^T, epilogue scatters Q(scaled), K, V^T
// block = 256 threads (8 waves), tile 64(M) x 128(N), K-chunks of 32,
// double-buffered: A via async engine, B (weights) via TDM
// ---------------------------------------------------------------------------
__global__ __launch_bounds__(256) void gemm_qkv(
    const _Float16* __restrict__ Xh, const _Float16* __restrict__ Wh,
    _Float16* __restrict__ Qh, _Float16* __restrict__ Kh,
    _Float16* __restrict__ Vth) {
  __shared__ __align__(16) _Float16 As[2][64][40];
  __shared__ __align__(16) _Float16 Bs[2][128][40];

  const int tid  = threadIdx.x;
  const int lane = tid & 31;
  const int wv   = tid >> 5;
  const int col  = lane & 15;
  const int hi   = lane >> 4;
  const int bm   = blockIdx.y * 64;
  const int bn   = blockIdx.x * 128;
  const int wm   = (wv >> 2) * 32;   // 0 or 32
  const int wn   = (wv & 3) * 32;    // 0..96
  const int kh8  = hi ? 8 : 0;
  const int kh16 = hi ? 16 : 0;

  const int arow = tid >> 2, aseg = (tid & 3) * 8;   // A: 64x32, 8 halves/thr
  const int brow = tid >> 1, bseg = (tid & 1) * 16;  // B fallback staging
  const unsigned ldsA0 = lds_addr_of(&As[0][0][0]);
  const unsigned ldsB0 = lds_addr_of(&Bs[0][0][0]);
  (void)brow; (void)bseg;

  auto stage = [&](int buf, int kc) {
    unsigned la = ldsA0 + (unsigned)buf * 64u * 40u * 2u +
                  (unsigned)(arow * 40 + aseg) * 2u;
    async_copy_b128(la, Xh + (size_t)(bm + arow) * CDIM + kc + aseg);
    unsigned lb = ldsB0 + (unsigned)buf * 128u * 40u * 2u;
#if HAVE_TDM
    if (wv == 0)
      tdm_load_tile_32xR(lb, Wh + (size_t)bn * CDIM + kc, 128u, (unsigned)CDIM);
#else
    unsigned lbt = lb + (unsigned)(brow * 40 + bseg) * 2u;
    const _Float16* srcB = Wh + (size_t)(bn + brow) * CDIM + kc + bseg;
    async_copy_b128(lbt, srcB);
    async_copy_b128(lbt + 16u, srcB + 8);
#endif
  };

  v8f acc[2][2];
#pragma unroll
  for (int i = 0; i < 2; ++i)
#pragma unroll
    for (int j = 0; j < 2; ++j) acc[i][j] = zero8();

  stage(0, 0);
  wait_async0();
  wait_tensor0();
  __syncthreads();

  int buf = 0;
  for (int kc = 0; kc < CDIM; kc += 32, buf ^= 1) {
    if (kc + 32 < CDIM) stage(buf ^ 1, kc + 32);

    v16h af[2], bf[2];
#pragma unroll
    for (int i = 0; i < 2; ++i)
      af[i] = frag_contig(&As[buf][wm + i * 16 + col][kh16]);
#pragma unroll
    for (int j = 0; j < 2; ++j)
      bf[j] = frag_contig(&Bs[buf][wn + j * 16 + col][kh16]);
#pragma unroll
    for (int i = 0; i < 2; ++i)
#pragma unroll
      for (int j = 0; j < 2; ++j) acc[i][j] = wmma_f16(af[i], bf[j], acc[i][j]);

    wait_async0();
    wait_tensor0();
    __syncthreads();
  }

  const float scale = 0.17677669529663687f;  // 1/sqrt(32)
#pragma unroll
  for (int i = 0; i < 2; ++i) {
#pragma unroll
    for (int j = 0; j < 2; ++j) {
      int gcol  = bn + wn + j * 16 + col;
      int which = gcol >> 9;            // 0=q,1=k,2=v
      int c     = gcol & (CDIM - 1);
      int head  = c >> 5, d = c & 31;
#pragma unroll
      for (int r = 0; r < 8; ++r) {
        int grow = bm + wm + i * 16 + hi * 8 + r;
        int bb   = grow >> 10, pos = grow & 1023;
        int bh   = bb * NHEADS + head;
        float v  = acc[i][j][r];
        if (which == 0)
          Qh[((size_t)bh * NPOS + pos) * HD + d] = (_Float16)(v * scale);
        else if (which == 1)
          Kh[((size_t)bh * NPOS + pos) * HD + d] = (_Float16)v;
        else
          Vth[((size_t)bh * HD + d) * NPOS + pos] = (_Float16)v;  // transposed
      }
    }
  }
}

// ---------------------------------------------------------------------------
// Flash attention: one wave per (b, h, 16-query tile). d=32, keys chunked by 32
// ---------------------------------------------------------------------------
__global__ __launch_bounds__(32) void attn_flash(
    const _Float16* __restrict__ Qh, const _Float16* __restrict__ Kh,
    const _Float16* __restrict__ Vth, _Float16* __restrict__ Ah) {
  __shared__ __align__(16) _Float16 Ps[16][32];  // P tile C-layout -> A-layout

  const int lane = threadIdx.x;
  const int col  = lane & 15;
  const int hi   = lane >> 4;
  const int kh8  = hi ? 8 : 0;
  const int kh16 = hi ? 16 : 0;
  const int blk  = blockIdx.x;       // (b*16+h)*64 + qt
  const int qt   = blk & 63;
  const int bh   = blk >> 6;
  const int b    = bh >> 4;
  const int h    = bh & 15;

  const _Float16* Qb = Qh  + (size_t)bh * NPOS * HD;
  const _Float16* Kb = Kh  + (size_t)bh * NPOS * HD;
  const _Float16* Vb = Vth + (size_t)bh * HD * NPOS;

  const v16h qa = frag_contig(Qb + (size_t)(qt * 16 + col) * HD + kh16);

  v8f o0 = zero8(), o1 = zero8();
  float mrow[8], lrow[8];
#pragma unroll
  for (int r = 0; r < 8; ++r) { mrow[r] = -3.0e38f; lrow[r] = 0.0f; }

  for (int kc = 0; kc < NPOS; kc += 32) {
    // prefetch next chunk of K and V into cache (global_prefetch_b8)
    if (kc + 32 < NPOS) {
      __builtin_prefetch(Kb + (size_t)(kc + 32 + col) * HD, 0, 1);
      __builtin_prefetch(Vb + (size_t)col * NPOS + kc + 32, 0, 1);
    }

    // scores: two 16x16 tiles, K-dim = d = 32 (one WMMA each)
    v16h kb0 = frag_contig(Kb + (size_t)(kc + col) * HD + kh16);
    v16h kb1 = frag_contig(Kb + (size_t)(kc + 16 + col) * HD + kh16);
    v8f s0 = wmma_f16(qa, kb0, zero8());
    v8f s1 = wmma_f16(qa, kb1, zero8());

    // online softmax per row (rows live across 16-lane halves)
#pragma unroll
    for (int r = 0; r < 8; ++r) {
      float rmax = fmaxf(s0[r], s1[r]);
#pragma unroll
      for (int off = 1; off < 16; off <<= 1)
        rmax = fmaxf(rmax, __shfl_xor(rmax, off, 32));
      float mnew  = fmaxf(mrow[r], rmax);
      float alpha = __expf(mrow[r] - mnew);
      float p0 = __expf(s0[r] - mnew);
      float p1 = __expf(s1[r] - mnew);
      float rsum = p0 + p1;
#pragma unroll
      for (int off = 1; off < 16; off <<= 1)
        rsum += __shfl_xor(rsum, off, 32);
      lrow[r] = lrow[r] * alpha + rsum;
      mrow[r] = mnew;
      o0[r] *= alpha;
      o1[r] *= alpha;
      int m = r + hi * 8;
      Ps[m][col]      = (_Float16)p0;
      Ps[m][16 + col] = (_Float16)p1;
    }
    __syncthreads();

    // P (16x32) as A-fragment; V^T rows give contiguous B-fragments
    v16h pa  = frag_contig(&Ps[col][kh16]);
    v16h vb0 = frag_contig(Vb + (size_t)col * NPOS + kc + kh16);
    v16h vb1 = frag_contig(Vb + (size_t)(16 + col) * NPOS + kc + kh16);
    o0 = wmma_f16(pa, vb0, o0);
    o1 = wmma_f16(pa, vb1, o1);
    __syncthreads();
  }

  // normalize and store attention output as f16 [b, pos, h*32 + d]
#pragma unroll
  for (int r = 0; r < 8; ++r) {
    int m = r + hi * 8;
    float inv = 1.0f / lrow[r];
    size_t row = ((size_t)(b * NPOS + qt * 16 + m)) * CDIM + h * HD;
    Ah[row + col]      = (_Float16)(o0[r] * inv);
    Ah[row + 16 + col] = (_Float16)(o1[r] * inv);
  }
}

// ---------------------------------------------------------------------------
// LePE 5x5 depthwise conv (from fp32 x), added into f16 attention output
// ---------------------------------------------------------------------------
__global__ void lepe_add(const float* __restrict__ x,
                         const float* __restrict__ wl,
                         const float* __restrict__ bl,
                         _Float16* __restrict__ Ah) {
  int idx = blockIdx.x * 256 + threadIdx.x;
  if (idx >= MROWS * CDIM) return;
  int c  = idx & (CDIM - 1);
  int t  = idx >> 9;
  int xw = t & 31;
  int y  = (t >> 5) & 31;
  int b  = t >> 10;
  float acc = bl[c];
#pragma unroll
  for (int ky = 0; ky < 5; ++ky) {
    int yy = y + ky - 2;
    if (yy < 0 || yy >= 32) continue;
#pragma unroll
    for (int kx = 0; kx < 5; ++kx) {
      int xx = xw + kx - 2;
      if (xx < 0 || xx >= 32) continue;
      acc += x[(((size_t)b * 32 + yy) * 32 + xx) * CDIM + c] *
             wl[c * 25 + ky * 5 + kx];
    }
  }
  Ah[idx] = (_Float16)((float)Ah[idx] + acc);
}

// ---------------------------------------------------------------------------
// Proj GEMM: [8192,512] x [512,512]^T + bias -> fp32 d_out
// ---------------------------------------------------------------------------
__global__ __launch_bounds__(256) void gemm_proj(
    const _Float16* __restrict__ Ahp, const _Float16* __restrict__ Wh,
    const float* __restrict__ bias, float* __restrict__ out) {
  __shared__ __align__(16) _Float16 As[2][64][40];
  __shared__ __align__(16) _Float16 Bs[2][128][40];

  const int tid  = threadIdx.x;
  const int lane = tid & 31;
  const int wv   = tid >> 5;
  const int col  = lane & 15;
  const int hi   = lane >> 4;
  const int bm   = blockIdx.y * 64;
  const int bn   = blockIdx.x * 128;
  const int wm   = (wv >> 2) * 32;
  const int wn   = (wv & 3) * 32;
  const int kh8  = hi ? 8 : 0;
  const int kh16 = hi ? 16 : 0;

  const int arow = tid >> 2, aseg = (tid & 3) * 8;
  const int brow = tid >> 1, bseg = (tid & 1) * 16;
  const unsigned ldsA0 = lds_addr_of(&As[0][0][0]);
  const unsigned ldsB0 = lds_addr_of(&Bs[0][0][0]);
  (void)brow; (void)bseg;

  auto stage = [&](int buf, int kc) {
    unsigned la = ldsA0 + (unsigned)buf * 64u * 40u * 2u +
                  (unsigned)(arow * 40 + aseg) * 2u;
    async_copy_b128(la, Ahp + (size_t)(bm + arow) * CDIM + kc + aseg);
    unsigned lb = ldsB0 + (unsigned)buf * 128u * 40u * 2u;
#if HAVE_TDM
    if (wv == 0)
      tdm_load_tile_32xR(lb, Wh + (size_t)bn * CDIM + kc, 128u, (unsigned)CDIM);
#else
    unsigned lbt = lb + (unsigned)(brow * 40 + bseg) * 2u;
    const _Float16* srcB = Wh + (size_t)(bn + brow) * CDIM + kc + bseg;
    async_copy_b128(lbt, srcB);
    async_copy_b128(lbt + 16u, srcB + 8);
#endif
  };

  v8f acc[2][2];
#pragma unroll
  for (int i = 0; i < 2; ++i)
#pragma unroll
    for (int j = 0; j < 2; ++j) acc[i][j] = zero8();

  stage(0, 0);
  wait_async0();
  wait_tensor0();
  __syncthreads();

  int buf = 0;
  for (int kc = 0; kc < CDIM; kc += 32, buf ^= 1) {
    if (kc + 32 < CDIM) stage(buf ^ 1, kc + 32);

    v16h af[2], bf[2];
#pragma unroll
    for (int i = 0; i < 2; ++i)
      af[i] = frag_contig(&As[buf][wm + i * 16 + col][kh16]);
#pragma unroll
    for (int j = 0; j < 2; ++j)
      bf[j] = frag_contig(&Bs[buf][wn + j * 16 + col][kh16]);
#pragma unroll
    for (int i = 0; i < 2; ++i)
#pragma unroll
      for (int j = 0; j < 2; ++j) acc[i][j] = wmma_f16(af[i], bf[j], acc[i][j]);

    wait_async0();
    wait_tensor0();
    __syncthreads();
  }

#pragma unroll
  for (int i = 0; i < 2; ++i) {
#pragma unroll
    for (int j = 0; j < 2; ++j) {
      int gcol = bn + wn + j * 16 + col;
      float bb = bias[gcol];
#pragma unroll
      for (int r = 0; r < 8; ++r) {
        int grow = bm + wm + i * 16 + hi * 8 + r;
        out[(size_t)grow * CDIM + gcol] = acc[i][j][r] + bb;
      }
    }
  }
}

// ---------------------------------------------------------------------------
extern "C" void kernel_launch(void* const* d_in, const int* in_sizes, int n_in,
                              void* d_out, int out_size, void* d_ws,
                              size_t ws_size, hipStream_t stream) {
  const float* x      = (const float*)d_in[0];
  const float* w_qkv  = (const float*)d_in[1];
  const float* w_proj = (const float*)d_in[2];
  const float* b_proj = (const float*)d_in[3];
  const float* w_lepe = (const float*)d_in[4];
  const float* b_lepe = (const float*)d_in[5];
  float* out = (float*)d_out;

  // workspace carve-up (f16 buffers), ~42 MB total
  _Float16* Xh     = (_Float16*)d_ws;
  _Float16* Wqkvh  = Xh    + (size_t)MROWS * CDIM;       // 8192*512
  _Float16* Wprojh = Wqkvh + (size_t)QKV_N * CDIM;       // 1536*512
  _Float16* Qh     = Wprojh + (size_t)CDIM * CDIM;       // 512*512
  _Float16* Kh     = Qh + (size_t)BATCH * NHEADS * NPOS * HD;
  _Float16* Vth    = Kh + (size_t)BATCH * NHEADS * NPOS * HD;
  _Float16* Ah     = Vth + (size_t)BATCH * NHEADS * NPOS * HD;

  const int nx = MROWS * CDIM;       // 4194304
  const int nq = QKV_N * CDIM;       // 786432
  const int np = CDIM * CDIM;        // 262144

  cvt_f32_to_f16<<<(nx + 255) / 256, 256, 0, stream>>>(x, Xh, nx);
  cvt_f32_to_f16<<<(nq + 255) / 256, 256, 0, stream>>>(w_qkv, Wqkvh, nq);
  cvt_f32_to_f16<<<(np + 255) / 256, 256, 0, stream>>>(w_proj, Wprojh, np);

  gemm_qkv<<<dim3(QKV_N / 128, MROWS / 64), 256, 0, stream>>>(
      Xh, Wqkvh, Qh, Kh, Vth);

  attn_flash<<<BATCH * NHEADS * (NPOS / 16), 32, 0, stream>>>(Qh, Kh, Vth, Ah);

  lepe_add<<<(nx + 255) / 256, 256, 0, stream>>>(x, w_lepe, b_lepe, Ah);

  gemm_proj<<<dim3(CDIM / 128, MROWS / 64), 256, 0, stream>>>(
      Ah, Wprojh, b_proj, out);
}
